// MinGRULM_1906965479606
// MI455X (gfx1250) — matrix operand order repaired and checked by reading, not compiled, not run on hardware
//
#include <hip/hip_runtime.h>
#include <hip/hip_bf16.h>

// ---------------- problem constants (match reference) ----------------
#define VV 32000
#define DD 1024
#define HH 1024
#define LL 6
#define KK 3
#define BB 4
#define SS 4096
#define MROWS (BB * SS)          // 16384 rows in every GEMM
#define NC 32                    // scan chunks
#define CH (SS / NC)             // 128 steps per chunk

typedef __bf16 bf16_t;
typedef __attribute__((ext_vector_type(16))) __bf16 v16bf;
typedef __attribute__((ext_vector_type(8)))  float  v8f;
typedef __attribute__((ext_vector_type(4)))  unsigned int u32x4;
typedef __attribute__((ext_vector_type(8)))  int i32x8;
typedef __attribute__((ext_vector_type(4)))  int i32x4;

static_assert(sizeof(v16bf) == 32, "v16bf must be 32 bytes");

union Frag32B { uint4 q[2]; v16bf v; };

#define HAVE_TDM __has_builtin(__builtin_amdgcn_tensor_load_to_lds)

// ---------------- embedding gather ----------------
__global__ __launch_bounds__(256) void embed_kernel(const int* __restrict__ x,
                                                    const float* __restrict__ emb,
                                                    float* __restrict__ h) {
  size_t i = (size_t)blockIdx.x * 256 + threadIdx.x;   // over B*S*D
  if (i >= (size_t)MROWS * DD) return;
  int d = (int)(i % DD);
  size_t bs = i / DD;
  int tok = x[bs];
  h[i] = emb[(size_t)tok * DD + d];
}

// ---------------- depthwise K=3 'same' conv (bf16 out: feeds GEMM) ----------------
__global__ __launch_bounds__(256) void dwconv_kernel(const float* __restrict__ x,
                                                     const float* __restrict__ w,  // [D][3]
                                                     const float* __restrict__ b,  // [D]
                                                     bf16_t* __restrict__ y) {
  size_t i = (size_t)blockIdx.x * 256 + threadIdx.x;   // over B*S*D
  if (i >= (size_t)MROWS * DD) return;
  int d = (int)(i % DD);
  size_t bs = i / DD;
  int s = (int)(bs % SS);
  int bb = (int)(bs / SS);
  const float* xb = x + ((size_t)bb * SS) * DD + d;
  float w0 = w[d * 3 + 0], w1 = w[d * 3 + 1], w2 = w[d * 3 + 2];
  float acc = b[d] + w1 * xb[(size_t)s * DD];
  if (s > 0)      acc += w0 * xb[(size_t)(s - 1) * DD];
  if (s + 1 < SS) acc += w2 * xb[(size_t)(s + 1) * DD];
  y[i] = (bf16_t)acc;
}

// ---------------- RMSNorm (bf16 out: feeds GEMM) ----------------
__global__ __launch_bounds__(256) void rmsnorm_kernel(const float* __restrict__ x,
                                                      const float* __restrict__ w,
                                                      bf16_t* __restrict__ y) {
  __shared__ float red[8];
  size_t row = blockIdx.x;
  const float* xr = x + row * DD;
  float s = 0.f;
  for (int d = threadIdx.x; d < DD; d += 256) { float v = xr[d]; s += v * v; }
  for (int off = 16; off > 0; off >>= 1) s += __shfl_down(s, off, 32);
  if ((threadIdx.x & 31) == 0) red[threadIdx.x >> 5] = s;
  __syncthreads();
  if (threadIdx.x == 0) {
    float t = 0.f;
    for (int i = 0; i < 8; ++i) t += red[i];
    red[0] = rsqrtf(t / (float)DD + 1.1920929e-7f);
  }
  __syncthreads();
  float r = red[0];
  bf16_t* yr = y + row * DD;
  for (int d = threadIdx.x; d < DD; d += 256) yr[d] = (bf16_t)(xr[d] * r * w[d]);
}

// ---------------- weight packers (fp32 -> bf16, [N][K] layout) ----------------
__global__ __launch_bounds__(256) void prep_w_kernel(const float* __restrict__ w,
                                                     bf16_t* __restrict__ o, int n) {
  int i = blockIdx.x * 256 + threadIdx.x;
  if (i < n) o[i] = (bf16_t)w[i];
}

// input [R][C] row-major -> output [C][R] (transpose)
__global__ __launch_bounds__(256) void prep_wt_kernel(const float* __restrict__ w,
                                                      bf16_t* __restrict__ o, int R, int C) {
  size_t i = (size_t)blockIdx.x * 256 + threadIdx.x;
  if (i >= (size_t)R * C) return;
  int r = (int)(i / C);
  int c = (int)(i % C);
  o[(size_t)c * R + r] = (bf16_t)w[i];
}

// ---------------- TDM 2D tile -> LDS (CDNA5 Tensor Data Mover) ----------------
#if HAVE_TDM
__device__ __forceinline__ void tdm_load_2d(unsigned lds_off, const bf16_t* gaddr,
                                            unsigned tile_k, unsigned tile_rows,
                                            unsigned row_stride_elems,
                                            unsigned tensor_w, unsigned tensor_h) {
  unsigned long long ga = (unsigned long long)(uintptr_t)gaddr;
  u32x4 g0;
  g0[0] = 1u;                                              // count=1 (valid), user mode
  g0[1] = lds_off;                                         // lds_addr (bytes)
  g0[2] = (unsigned)(ga & 0xffffffffu);                    // global_addr[31:0]
  g0[3] = (unsigned)((ga >> 32) & 0x01ffffffu) | (2u << 30); // addr[56:32] | type=2
  i32x8 g1;
  g1[0] = (int)(1u << 16);                                 // wg_mask=0, data_size=1 (2B)
  g1[1] = (int)((tensor_w & 0xffffu) << 16);               // tensor_dim0[15:0]
  g1[2] = (int)((tensor_w >> 16) & 0xffffu) | (int)((tensor_h & 0xffffu) << 16);
  g1[3] = (int)((tensor_h >> 16) & 0xffffu) | (int)((tile_k & 0xffffu) << 16); // tile_dim0
  g1[4] = (int)(tile_rows & 0xffffu);                      // tile_dim1 (tile_dim2=0)
  g1[5] = (int)row_stride_elems;                           // tensor_dim0_stride[31:0]
  g1[6] = 0;                                               // stride[47:32] | dim1_stride lo
  g1[7] = 0;
  i32x4 z4 = {0, 0, 0, 0};
  i32x8 z8 = {0, 0, 0, 0, 0, 0, 0, 0};
  // 6-arg signature on this toolchain: (g0, g1, g2, g3, g4, cpol)
  __builtin_amdgcn_tensor_load_to_lds(g0, g1, z4, z4, z8, 0);
}
#endif

// ---------------- fused WMMA bf16 GEMM ----------------
// C = A[M][K](bf16) @ Bp^T, Bp stored [N][K] bf16.
// mode 0: +bias -> fp32 C ; mode 1: +bias +resid -> fp32 C ; mode 2: +bias, GELU -> bf16 Cb
#define TM 128
#define TN 64
#define TKB 64                   // K-tile (two WMMA K-steps of 32)

__global__ __launch_bounds__(256) void gemm_bf16_kernel(
    const bf16_t* __restrict__ A, const bf16_t* __restrict__ Bp,
    float* __restrict__ C, bf16_t* __restrict__ Cb,
    const float* __restrict__ bias, const float* __restrict__ resid,
    int M, int N, int K, int mode) {
  __shared__ __align__(16) bf16_t As[2][TM * TKB];   // 2 x 16KB, row stride 128B
  __shared__ __align__(16) bf16_t Bs[2][TN * TKB];   // 2 x 8KB,  row stride 128B

  const int tid  = threadIdx.x;
  const int lane = tid & 31;
  const int wid  = tid >> 5;
  const int wm   = (wid & 3) * 32;
  const int wn   = (wid >> 2) * 32;
  const int bm   = blockIdx.y * TM;
  const int bn   = blockIdx.x * TN;
  const int half = lane >> 4;
  const int l15  = lane & 15;

  const bf16_t* Ab = A + (size_t)bm * K;
  const bf16_t* Bb = Bp + (size_t)bn * K;

  v8f acc[2][2];
#pragma unroll
  for (int i = 0; i < 2; ++i)
#pragma unroll
    for (int j = 0; j < 2; ++j) acc[i][j] = {};

  const int nk = K / TKB;

#if HAVE_TDM
  const unsigned asOff0 = (unsigned)(uintptr_t)(void*)&As[0][0];
  const unsigned asOff1 = (unsigned)(uintptr_t)(void*)&As[1][0];
  const unsigned bsOff0 = (unsigned)(uintptr_t)(void*)&Bs[0][0];
  const unsigned bsOff1 = (unsigned)(uintptr_t)(void*)&Bs[1][0];
  if (wid == 0) {   // one wave drives the DMA engine
    tdm_load_2d(asOff0, Ab, TKB, TM, K, K, TM);
    tdm_load_2d(bsOff0, Bb, TKB, TN, K, K, TN);
  }
#endif

  for (int t = 0; t < nk; ++t) {
    const int cur = t & 1;
#if HAVE_TDM
    if (wid == 0) {
      if (t + 1 < nk) {
        unsigned ao = (cur == 0) ? asOff1 : asOff0;
        unsigned bo = (cur == 0) ? bsOff1 : bsOff0;
        tdm_load_2d(ao, Ab + (size_t)(t + 1) * TKB, TKB, TM, K, K, TM);
        tdm_load_2d(bo, Bb + (size_t)(t + 1) * TKB, TKB, TN, K, K, TN);
        __builtin_amdgcn_s_wait_tensorcnt(2);   // tile t finished (in-order)
      } else {
        __builtin_amdgcn_s_wait_tensorcnt(0);
      }
    }
    __syncthreads();
#else
    // fallback: cooperative bf16 tile copy (no conversion needed)
#pragma unroll
    for (int it = 0; it < 4; ++it) {           // A: 128x64 bf16 = 1024 x 16B
      int idx = tid + it * 256;
      int r = idx >> 3, c8 = idx & 7;
      *(uint4*)&As[cur][r * TKB + c8 * 8] =
          *(const uint4*)(Ab + (size_t)r * K + (size_t)t * TKB + c8 * 8);
    }
#pragma unroll
    for (int it = 0; it < 2; ++it) {           // B: 64x64 bf16 = 512 x 16B
      int idx = tid + it * 256;
      int r = idx >> 3, c8 = idx & 7;
      *(uint4*)&Bs[cur][r * TKB + c8 * 8] =
          *(const uint4*)(Bb + (size_t)r * K + (size_t)t * TKB + c8 * 8);
    }
    __syncthreads();
#endif

#pragma unroll
    for (int ks = 0; ks < 2; ++ks) {
      const int kk = ks * 32;
      v16bf afrag[2], bfrag[2];
#pragma unroll
      for (int i = 0; i < 2; ++i) {
        const bf16_t* ap = &As[cur][(wm + i * 16 + l15) * TKB + kk];
        Frag32B fa;
        fa.q[0] = *(const uint4*)(ap + half * 8);
        fa.q[1] = *(const uint4*)(ap + 16 + half * 8);
        afrag[i] = fa.v;
      }
#pragma unroll
      for (int j = 0; j < 2; ++j) {
        const bf16_t* bp = &Bs[cur][(wn + j * 16 + l15) * TKB + kk + half * 8];
        Frag32B fb;
        fb.q[0] = *(const uint4*)(bp);
        fb.q[1] = *(const uint4*)(bp + 16);
        bfrag[j] = fb.v;
      }
#pragma unroll
      for (int i = 0; i < 2; ++i)
#pragma unroll
        for (int j = 0; j < 2; ++j)
          acc[i][j] = __builtin_amdgcn_wmma_f32_16x16x32_bf16(
              false, afrag[i], false, bfrag[j], (short)0, acc[i][j], false, false);
    }
    __syncthreads();
  }

  // ---- fused epilogue ----
#pragma unroll
  for (int i = 0; i < 2; ++i) {
#pragma unroll
    for (int j = 0; j < 2; ++j) {
      int n = bn + wn + j * 16 + l15;
      float bv = bias ? bias[n] : 0.f;
#pragma unroll
      for (int r = 0; r < 8; ++r) {
        int m = bm + wm + i * 16 + r + half * 8;
        float v = acc[i][j][r] + bv;
        if (mode == 1) v += resid[(size_t)m * N + n];
        if (mode == 2) v = 0.5f * v * (1.0f + erff(v * 0.70710678118f));
        if (Cb) Cb[(size_t)m * N + n] = (bf16_t)v;
        else    C[(size_t)m * N + n] = v;
      }
    }
  }
}

// ---------------- chunked parallel scan for MinGRU ----------------
__device__ __forceinline__ float sigf(float v) { return 1.f / (1.f + __expf(-v)); }
__device__ __forceinline__ float gfun(float v) { return v >= 0.f ? v + 0.5f : sigf(v); }

__global__ __launch_bounds__(256) void scan_pass1(const float* __restrict__ kb,
                                                  const float* __restrict__ tb,
                                                  float* __restrict__ a,
                                                  float* __restrict__ bp) {
  int tid = blockIdx.x * 256 + threadIdx.x;
  if (tid >= BB * DD * NC) return;
  int d = tid % DD;
  int c = (tid / DD) % NC;
  int b = tid / (DD * NC);
  size_t base = ((size_t)b * SS + (size_t)c * CH) * DD + d;
  float av = 1.f, hv = 0.f;
  for (int s = 0; s < CH; ++s) {
    size_t idx = base + (size_t)s * DD;
    float z  = sigf(kb[idx]);
    float gh = gfun(tb[idx]);
    av *= (1.f - z);
    hv = hv + z * (gh - hv);
  }
  size_t ai = ((size_t)b * NC + c) * DD + d;
  a[ai]  = av;
  bp[ai] = hv;
}

__global__ __launch_bounds__(256) void scan_pass2(const float* __restrict__ a,
                                                  const float* __restrict__ bp,
                                                  const float* __restrict__ hprev,
                                                  float* __restrict__ hstart) {
  int tid = blockIdx.x * 256 + threadIdx.x;
  if (tid >= BB * DD) return;
  int d = tid % DD;
  int b = tid / DD;
  float hc = gfun(hprev[tid]);
  for (int c = 0; c < NC; ++c) {
    size_t i = ((size_t)b * NC + c) * DD + d;
    hstart[i] = hc;
    hc = a[i] * hc + bp[i];
  }
}

__global__ __launch_bounds__(256) void scan_pass3(const float* __restrict__ kb,
                                                  const float* __restrict__ tb,
                                                  const float* __restrict__ hstart,
                                                  float* __restrict__ h) {
  int tid = blockIdx.x * 256 + threadIdx.x;
  if (tid >= BB * DD * NC) return;
  int d = tid % DD;
  int c = (tid / DD) % NC;
  int b = tid / (DD * NC);
  float hc = hstart[((size_t)b * NC + c) * DD + d];
  size_t base = ((size_t)b * SS + (size_t)c * CH) * DD + d;
  for (int s = 0; s < CH; ++s) {
    size_t idx = base + (size_t)s * DD;
    float z  = sigf(kb[idx]);
    float gh = gfun(tb[idx]);
    hc = hc + z * (gh - hc);
    h[idx] += hc;
  }
}

// ---------------- launcher ----------------
extern "C" void kernel_launch(void* const* d_in, const int* in_sizes, int n_in,
                              void* d_out, int out_size, void* d_ws, size_t ws_size,
                              hipStream_t stream) {
  const int*   x         = (const int*)d_in[0];
  const float* h_prev    = (const float*)d_in[1];
  const float* emb       = (const float*)d_in[2];
  const float* conv_dw_w = (const float*)d_in[3];
  const float* conv_dw_b = (const float*)d_in[4];
  const float* conv_pw_w = (const float*)d_in[5];
  const float* conv_pw_b = (const float*)d_in[6];
  const float* norm1_w   = (const float*)d_in[7];
  const float* norm2_w   = (const float*)d_in[8];
  const float* zw        = (const float*)d_in[9];
  const float* zb        = (const float*)d_in[10];
  const float* hw        = (const float*)d_in[11];
  const float* hb        = (const float*)d_in[12];
  const float* f1w       = (const float*)d_in[13];
  const float* f1b       = (const float*)d_in[14];
  const float* f2w       = (const float*)d_in[15];
  const float* f2b       = (const float*)d_in[16];
  const float* normf_w   = (const float*)d_in[17];
  const float* out_w     = (const float*)d_in[18];
  const float* out_b     = (const float*)d_in[19];
  float* out = (float*)d_out;

  // workspace layout
  const size_t ACT = (size_t)MROWS * DD;
  char* ws = (char*)d_ws;
  float*  h    = (float*)(ws);                               // residual stream (fp32)
  float*  kb   = (float*)(ws + 1 * ACT * sizeof(float));     // gate pre-act
  float*  tb   = (float*)(ws + 2 * ACT * sizeof(float));     // htilde pre-act
  bf16_t* nbf  = (bf16_t*)(ws + 3 * ACT * sizeof(float));    // bf16 GEMM input
  bf16_t* kbf  = (bf16_t*)(ws + 3 * ACT * sizeof(float) + ACT * sizeof(bf16_t));
  char*   p1   = ws + 3 * ACT * sizeof(float) + 2 * ACT * sizeof(bf16_t);
  bf16_t* wbuf = (bf16_t*)(p1);                              // packed weights (2MB)
  char*   p2   = p1 + (size_t)DD * HH * sizeof(bf16_t);
  float*  sa   = (float*)(p2);
  float*  sb   = (float*)(p2 + (size_t)BB * DD * NC * sizeof(float));
  float*  shs  = (float*)(p2 + 2 * (size_t)BB * DD * NC * sizeof(float));

  const int elemBlocks = (int)(((size_t)MROWS * DD + 255) / 256);
  const int wBlocks    = (DD * DD + 255) / 256;
  const int wtBlocks   = (DD * HH + 255) / 256;
  const dim3 gemmGridD(DD / TN, MROWS / TM);
  const dim3 gemmGridH(HH / TN, MROWS / TM);
  const int scanBlocks1 = (BB * DD * NC + 255) / 256;
  const int scanBlocks2 = (BB * DD + 255) / 256;

  embed_kernel<<<elemBlocks, 256, 0, stream>>>(x, emb, h);

  for (int l = 0; l < LL; ++l) {
    // --- depthwise conv + pointwise proj (+bias +residual) ---
    dwconv_kernel<<<elemBlocks, 256, 0, stream>>>(
        h, conv_dw_w + (size_t)l * DD * KK, conv_dw_b + (size_t)l * DD, nbf);
    prep_w_kernel<<<wBlocks, 256, 0, stream>>>(conv_pw_w + (size_t)l * DD * DD, wbuf, DD * DD);
    gemm_bf16_kernel<<<gemmGridD, 256, 0, stream>>>(
        nbf, wbuf, h, nullptr, conv_pw_b + (size_t)l * DD, h, MROWS, DD, DD, 1);

    // --- MinGRU block ---
    rmsnorm_kernel<<<MROWS, 256, 0, stream>>>(h, norm1_w + (size_t)l * DD, nbf);
    prep_w_kernel<<<wBlocks, 256, 0, stream>>>(zw + (size_t)l * DD * DD, wbuf, DD * DD);
    gemm_bf16_kernel<<<gemmGridD, 256, 0, stream>>>(
        nbf, wbuf, kb, nullptr, zb + (size_t)l * DD, nullptr, MROWS, DD, DD, 0);
    prep_w_kernel<<<wBlocks, 256, 0, stream>>>(hw + (size_t)l * DD * DD, wbuf, DD * DD);
    gemm_bf16_kernel<<<gemmGridD, 256, 0, stream>>>(
        nbf, wbuf, tb, nullptr, hb + (size_t)l * DD, nullptr, MROWS, DD, DD, 0);
    scan_pass1<<<scanBlocks1, 256, 0, stream>>>(kb, tb, sa, sb);
    scan_pass2<<<scanBlocks2, 256, 0, stream>>>(sa, sb, h_prev, shs);
    scan_pass3<<<scanBlocks1, 256, 0, stream>>>(kb, tb, shs, h);

    // --- FFN block ---
    rmsnorm_kernel<<<MROWS, 256, 0, stream>>>(h, norm2_w + (size_t)l * DD, nbf);
    prep_wt_kernel<<<wtBlocks, 256, 0, stream>>>(f1w + (size_t)l * DD * HH, wbuf, DD, HH);
    gemm_bf16_kernel<<<gemmGridH, 256, 0, stream>>>(
        nbf, wbuf, nullptr, kbf, f1b + (size_t)l * HH, nullptr, MROWS, HH, DD, 2); // GELU->bf16
    prep_wt_kernel<<<wtBlocks, 256, 0, stream>>>(f2w + (size_t)l * HH * DD, wbuf, HH, DD);
    gemm_bf16_kernel<<<gemmGridD, 256, 0, stream>>>(
        kbf, wbuf, h, nullptr, f2b + (size_t)l * DD, h, MROWS, DD, HH, 1);
  }

  // --- final norm + projection ---
  rmsnorm_kernel<<<MROWS, 256, 0, stream>>>(h, normf_w, nbf);
  prep_wt_kernel<<<wtBlocks, 256, 0, stream>>>(out_w, wbuf, DD, HH);
  gemm_bf16_kernel<<<gemmGridH, 256, 0, stream>>>(
      nbf, wbuf, out, nullptr, out_b, nullptr, MROWS, HH, DD, 0);
}
